// TICLNet_15264313770591
// MI455X (gfx1250) — matrix operand order repaired and checked by reading, not compiled, not run on hardware
//
#include <hip/hip_runtime.h>
#include <hip/hip_bf16.h>

typedef __attribute__((ext_vector_type(16))) _Float16 v16h;
typedef __attribute__((ext_vector_type(8)))  _Float16 v8h;
typedef __attribute__((ext_vector_type(2)))  _Float16 v2h;
typedef __attribute__((ext_vector_type(8)))  float    v8f;
typedef __attribute__((ext_vector_type(4)))  float    v4f;

#define TB_  4096
#define TL_  50
#define TT_  64

__device__ __forceinline__ float fast_exp2(float x) {
#if __has_builtin(__builtin_amdgcn_exp2f)
    return __builtin_amdgcn_exp2f(x);
#else
    return exp2f(x);
#endif
}
__device__ __forceinline__ float fast_rcp(float x) {
#if __has_builtin(__builtin_amdgcn_rcpf)
    return __builtin_amdgcn_rcpf(x);
#else
    return 1.0f / x;
#endif
}
__device__ __forceinline__ float sigm(float x) {
    return fast_rcp(1.0f + fast_exp2(-1.44269504f * x));
}
__device__ __forceinline__ float tanha(float x) {
#if __has_builtin(__builtin_amdgcn_tanhf)
    return __builtin_amdgcn_tanhf(x);
#else
    x = fminf(fmaxf(x, -15.0f), 15.0f);
    const float e = fast_exp2(2.88539008f * x);
    return (e - 1.0f) * fast_rcp(e + 1.0f);
#endif
}

// U row layout (K = 0..31): [x0 x1 x2 | 0 | h0..h19 | 1.0 | 0 x7]
//   K<3: x (register-inserted, LDS slots stay 0)   K==3: pad
//   4<=K<24: h[K-4]                                K==24: bias column (1.0)
__global__ __launch_bounds__(128) void ticlnet_fused(
    const float* __restrict__ x,     const int*   __restrict__ lengths,
    const float* __restrict__ w_ih,  const float* __restrict__ w_hh,
    const float* __restrict__ b_ih,  const float* __restrict__ b_hh,
    const float* __restrict__ w_lin, const float* __restrict__ b_lin,
    const float* __restrict__ c1_w,  const float* __restrict__ c1_b,
    const float* __restrict__ c2_w,  const float* __restrict__ c2_b,
    const float* __restrict__ c3_w,  const float* __restrict__ c3_b,
    const float* __restrict__ fc1_w, const float* __restrict__ fc1_b,
    const float* __restrict__ fc2_w, const float* __restrict__ fc2_b,
    float* __restrict__ out)
{
    // per-wave LSTM scratch (wave-private slices; same-wave LDS ops are in-order)
    __shared__ float    s_gatesT[4][80 * 16];  // transposed gates [col][row] per wave
    __shared__ _Float16 s_u[4][16 * 32];       // padded U rows, f16, per wave
    // conv-phase scratch (used after __syncthreads)
    __shared__ float    s_feat[16][50];
    __shared__ float    s_y1[32][46];
    __shared__ float    s_y2[32][22];
    __shared__ float    s_y3[32 * 19];
    __shared__ float    s_fc[16];

    const int tid  = threadIdx.x;
    const int wv   = tid >> 5;      // wave in block (0..3)
    const int lane = tid & 31;
    const int hi   = lane >> 4;     // lane half
    const int n    = lane & 15;     // row (A) or column (B/C) index
    const int b    = blockIdx.x;

    // ---------------- B fragments: Wpad[32 x 80], bias folded at K==24 ----
    // B layout (16-bit, 32x16): lane = hi*16 + n holds column n, element e -> K = 8*hi + e (e<8), 16 + 8*hi + (e-8) (e>=8)
    v16h bfrag[5];
    #pragma unroll
    for (int ct = 0; ct < 5; ++ct) {
        const int col = ct * 16 + n;          // gate column 0..79
        #pragma unroll
        for (int e = 0; e < 16; ++e) {
            const int K = (e < 8) ? (hi * 8 + e) : (16 + hi * 8 + (e - 8));
            float wvv;
            if (K < 3)        wvv = w_ih[col * 3 + K];
            else if (K == 3)  wvv = 0.0f;
            else if (K < 24)  wvv = w_hh[col * 20 + (K - 4)];
            else if (K == 24) wvv = b_ih[col] + b_hh[col];
            else              wvv = 0.0f;
            bfrag[ct][e] = (_Float16)wvv;
        }
    }

    // ---------------- init U rows + per-lane state ----------------
    v8h* urow = (v8h*)&s_u[wv][n * 32];
    {
        const v8h z8 = (v8h)(_Float16)0.0f;
        urow[hi]     = z8;                    // slots 0..7 / 8..15
        urow[2 + hi] = z8;                    // slots 16..23 / 24..31
        if (hi == 0) s_u[wv][n * 32 + 24] = (_Float16)1.0f;  // bias column
    }
    const int ubase = hi * 10;                // this lane updates units [ubase, ubase+10)
    float c_st[10], h_st[10];
    #pragma unroll
    for (int j = 0; j < 10; ++j) { c_st[j] = 0.0f; h_st[j] = 0.0f; }

    const int  ll    = wv * 16 + n;           // layer index in batch (0..63)
    const bool valid = (ll < TL_);
    const int  seq   = b * TL_ + ll;
    const int  len   = valid ? lengths[seq] : 0;
    const float* xp  = x + (size_t)(valid ? seq : 0) * (TT_ * 3);

    const v8f zacc = {0.f, 0.f, 0.f, 0.f, 0.f, 0.f, 0.f, 0.f};

    // ---------------- Timestep loop ----------------
    for (int t = 0; t < TT_; ++t) {
        // all lanes load (hi=1 lanes hit the same lines -> coalesced), insert branch-free
        const float xr0 = xp[t * 3 + 0];
        const float xr1 = xp[t * 3 + 1];
        const float xr2 = xp[t * 3 + 2];

        // A fragment: two contiguous 8-half runs -> 2x ds_load_b128
        v8h alo = urow[hi];                   // K = 8*hi .. 8*hi+7
        const v8h ahi2 = urow[2 + hi];        // K = 16+8*hi .. +7
        alo[0] = hi ? alo[0] : (_Float16)xr0; // K=0 (hi=0 only)
        alo[1] = hi ? alo[1] : (_Float16)xr1; // K=1
        alo[2] = hi ? alo[2] : (_Float16)xr2; // K=2
        v16h a;
        #pragma unroll
        for (int e = 0; e < 8; ++e) { a[e] = alo[e]; a[8 + e] = ahi2[e]; }

        // gates[16x80] = A x Wpad (5 column tiles), f32 accumulate, C = 0
        v8f d[5];
        #pragma unroll
        for (int ct = 0; ct < 5; ++ct)
            d[ct] = __builtin_amdgcn_wmma_f32_16x16x32_f16(
                        false, a, false, bfrag[ct], (short)0, zacc, false, false);

        // transposed spill: rows are contiguous -> 2x ds_store_b128 per tile
        #pragma unroll
        for (int ct = 0; ct < 5; ++ct) {
            float* gp = &s_gatesT[wv][(ct * 16 + n) * 16 + 8 * hi];
            v4f lo4 = {d[ct][0], d[ct][1], d[ct][2], d[ct][3]};
            v4f hi4 = {d[ct][4], d[ct][5], d[ct][6], d[ct][7]};
            *(v4f*)(gp)     = lo4;
            *(v4f*)(gp + 4) = hi4;
        }

        // branch-free LSTM cell update (2 units per iter, paired f16 h store)
        const bool act = (t < len);
        const float* gT = s_gatesT[wv];
        #pragma unroll
        for (int j = 0; j < 5; ++j) {
            const int u0 = ubase + 2 * j;
            const float i0 = gT[(u0)      * 16 + n], i1 = gT[(u0 + 1)  * 16 + n];
            const float f0 = gT[(20 + u0) * 16 + n], f1 = gT[(21 + u0) * 16 + n];
            const float g0 = gT[(40 + u0) * 16 + n], g1 = gT[(41 + u0) * 16 + n];
            const float o0 = gT[(60 + u0) * 16 + n], o1 = gT[(61 + u0) * 16 + n];
            const float cn0 = sigm(f0) * c_st[2 * j]     + sigm(i0) * tanha(g0);
            const float cn1 = sigm(f1) * c_st[2 * j + 1] + sigm(i1) * tanha(g1);
            const float hn0 = sigm(o0) * tanha(cn0);
            const float hn1 = sigm(o1) * tanha(cn1);
            c_st[2 * j]     = act ? cn0 : c_st[2 * j];
            c_st[2 * j + 1] = act ? cn1 : c_st[2 * j + 1];
            h_st[2 * j]     = act ? hn0 : h_st[2 * j];
            h_st[2 * j + 1] = act ? hn1 : h_st[2 * j + 1];
            v2h hp;
            hp[0] = (_Float16)h_st[2 * j];
            hp[1] = (_Float16)h_st[2 * j + 1];
            *(v2h*)(&s_u[wv][n * 32 + 4 + u0]) = hp;   // 4B aligned
        }
    }

    // ---------------- Layer summary: sigmoid(h @ w_lin^T + b_lin), 0 if len==0 ----
    if (valid) {
        #pragma unroll
        for (int oi = 0; oi < 8; ++oi) {
            const int o = hi * 8 + oi;
            float acc = b_lin[o];
            #pragma unroll
            for (int u = 0; u < 20; ++u)
                acc += (float)s_u[wv][n * 32 + 4 + u] * w_lin[o * 20 + u];
            s_feat[o][ll] = (len > 0) ? sigm(acc) : 0.0f;
        }
    }
    __syncthreads();

    // ---------------- Conv1: 16x50 -> 32x46 (k5,s1) ----------------
    for (int idx = tid; idx < 32 * 46; idx += 128) {
        const int oc = idx / 46, p = idx % 46;
        float acc = c1_b[oc];
        #pragma unroll
        for (int ic = 0; ic < 16; ++ic)
            #pragma unroll
            for (int k = 0; k < 5; ++k)
                acc += s_feat[ic][p + k] * c1_w[oc * 80 + ic * 5 + k];
        s_y1[oc][p] = fmaxf(acc, 0.0f);
    }
    __syncthreads();

    // ---------------- Conv2: 32x46 -> 32x22 (k4,s2) ----------------
    for (int idx = tid; idx < 32 * 22; idx += 128) {
        const int oc = idx / 22, p = idx % 22;
        float acc = c2_b[oc];
        #pragma unroll
        for (int ic = 0; ic < 32; ++ic)
            #pragma unroll
            for (int k = 0; k < 4; ++k)
                acc += s_y1[ic][2 * p + k] * c2_w[oc * 128 + ic * 4 + k];
        s_y2[oc][p] = fmaxf(acc, 0.0f);
    }
    __syncthreads();

    // ---------------- Conv3: 32x22 -> 32x19 (k4,s1) ----------------
    for (int idx = tid; idx < 32 * 19; idx += 128) {
        const int oc = idx / 19, p = idx % 19;
        float acc = c3_b[oc];
        #pragma unroll
        for (int ic = 0; ic < 32; ++ic)
            #pragma unroll
            for (int k = 0; k < 4; ++k)
                acc += s_y2[ic][p + k] * c3_w[oc * 128 + ic * 4 + k];
        s_y3[oc * 19 + p] = fmaxf(acc, 0.0f);
    }
    __syncthreads();

    // ---------------- FC1: 608 -> 16, relu ----------------
    if (tid < 16) {
        float acc = fc1_b[tid];
        for (int j = 0; j < 32 * 19; ++j)
            acc += s_y3[j] * fc1_w[tid * 608 + j];
        s_fc[tid] = fmaxf(acc, 0.0f);
    }
    __syncthreads();

    // ---------------- FC2: 16 -> 4 ----------------
    if (tid < 4) {
        float acc = fc2_b[tid];
        #pragma unroll
        for (int j = 0; j < 16; ++j)
            acc += s_fc[j] * fc2_w[tid * 16 + j];
        out[b * 4 + tid] = acc;
    }
}

extern "C" void kernel_launch(void* const* d_in, const int* in_sizes, int n_in,
                              void* d_out, int out_size, void* d_ws, size_t ws_size,
                              hipStream_t stream) {
    const float* x      = (const float*)d_in[0];
    const int*   lens   = (const int*)  d_in[1];
    const float* w_ih   = (const float*)d_in[2];
    const float* w_hh   = (const float*)d_in[3];
    const float* b_ih   = (const float*)d_in[4];
    const float* b_hh   = (const float*)d_in[5];
    const float* w_lin  = (const float*)d_in[6];
    const float* b_lin  = (const float*)d_in[7];
    const float* c1_w   = (const float*)d_in[8];
    const float* c1_b   = (const float*)d_in[9];
    const float* c2_w   = (const float*)d_in[10];
    const float* c2_b   = (const float*)d_in[11];
    const float* c3_w   = (const float*)d_in[12];
    const float* c3_b   = (const float*)d_in[13];
    const float* fc1_w  = (const float*)d_in[14];
    const float* fc1_b  = (const float*)d_in[15];
    const float* fc2_w  = (const float*)d_in[16];
    const float* fc2_b  = (const float*)d_in[17];
    float* outp = (float*)d_out;

    ticlnet_fused<<<TB_, 128, 0, stream>>>(
        x, lens, w_ih, w_hh, b_ih, b_hh, w_lin, b_lin,
        c1_w, c1_b, c2_w, c2_b, c3_w, c3_b,
        fc1_w, fc1_b, fc2_w, fc2_b, outp);
}
